// StackedLSTM_88974542504149
// MI455X (gfx1250) — matrix operand order repaired and checked by reading, not compiled, not run on hardware
//
#include <hip/hip_runtime.h>

typedef __attribute__((ext_vector_type(16))) _Float16 v16h;
typedef __attribute__((ext_vector_type(8)))  float    v8f;

#define DEVINL __device__ __forceinline__

namespace {
constexpr int kB = 2048;   // batch
constexpr int kT = 2048;   // timesteps
constexpr int kD = 6;      // input dim (== H, so layer1 uses same loader)
constexpr int kH = 6;      // hidden dim
constexpr int kG = 24;     // 4*H gates
}

// Physical gate row -> logical index into the PyTorch-order [i f g o] 4H vector.
// Chosen so tile0 row pairs (v, v+8) share an activation type (all sigmoid) and
// tile1 rows 16..23 are {o4,o5,g0..g5} (g last 6 -> packed tanh).
__device__ const int kPerm[24] = {
   0, 1, 2,  6, 7, 8, 18, 19,   // tile0 rows 0-7 : i0 i1 i2 f0 f1 f2 o0 o1
   3, 4, 5,  9,10,11, 20, 21,   // tile0 rows 8-15: i3 i4 i5 f3 f4 f5 o2 o3
  22,23,12,13,14,15,16,17 };    // tile1 rows16-23: o4 o5 g0 g1 g2 g3 g4 g5

DEVINL float fast_sigmoid(float x) {
  // 1 / (1 + 2^(-x*log2 e))  -> v_mul, v_exp_f32, v_add, v_rcp_f32
  return __builtin_amdgcn_rcpf(1.0f + __builtin_amdgcn_exp2f(-1.44269504f * x));
}
DEVINL float fast_tanh(float x) {
  // 1 - 2/(e^{2x}+1); e^{2x} = 2^{x * 2*log2 e}. tanh(0) == 0 exactly
  // (exp2(0)=1, rcp(2)=0.5) which the upper-half zero-state invariant relies on.
  float t = __builtin_amdgcn_exp2f(2.88539008f * x);
  return 1.0f - 2.0f * __builtin_amdgcn_rcpf(t + 1.0f);
}
DEVINL float swap16(float v) {
  // ds_swizzle SWAPX16: offset = {xor=0x10, or=0, and=0x1f} = 0x401F
  return __int_as_float(__builtin_amdgcn_ds_swizzle(__float_as_int(v), 0x401F));
}

// A operand (Wcat = [W_ih | W_hh], 24x12, K zero-padded to 32) in the 16-bit
// 16x32 A operand, contiguous k order (as B): lane m<16 holds row m K=0..15 (elements 0..15);
// lane m+16 holds row m K=16..31 (=0). tile1 rows >=24 are zero.
DEVINL v16h load_A(const float* Wih, const float* Whh, int tile, int lane) {
  v16h a;
  #pragma unroll
  for (int i = 0; i < 16; ++i) a[i] = (_Float16)0.0f;
  int r  = tile * 16 + (lane & 15);
  int kb = (lane < 16) ? 0 : 16;
  if (r < kG) {
    int lg = kPerm[r];
    #pragma unroll
    for (int k = 0; k < 16; ++k) {
      int kk = kb + k;
      float w = 0.0f;
      if (kk < kD)            w = Wih[lg * kD + kk];
      else if (kk < kD + kH)  w = Whh[lg * kH + (kk - kD)];
      a[k] = (_Float16)w;
    }
  }
  return a;
}

// Bias as the WMMA C operand (f32 16x16 C layout: VGPR v = row v (lanes 0-15)
// or row v+8 (lanes 16-31)). tile1 upper rows 24..31 = 0 (pad).
DEVINL v8f load_bias(const float* bih, const float* bhh, int tile, int lane) {
  v8f c;
  #pragma unroll
  for (int v = 0; v < 8; ++v) {
    int r = tile * 16 + v + ((lane < 16) ? 0 : 8);
    c[v] = (r < kG) ? (bih[kPerm[r]] + bhh[kPerm[r]]) : 0.0f;
  }
  return c;
}

// One LSTM cell update from the two gate tiles. Low lanes carry real state;
// upper lanes provably keep h==c==0 (their g gates come from zero pad rows).
DEVINL void step_update(v8f d0, v8f d1, float (&h)[6], float (&c)[6]) {
  #pragma unroll
  for (int v = 0; v < 8; ++v) d0[v] = fast_sigmoid(d0[v]);   // rows 0-15: i,f,o
  d1[0] = fast_sigmoid(d1[0]);                               // o4
  d1[1] = fast_sigmoid(d1[1]);                               // o5
  #pragma unroll
  for (int v = 2; v < 8; ++v) d1[v] = fast_tanh(d1[v]);      // g0..g5

  float up[8];
  #pragma unroll
  for (int v = 0; v < 8; ++v) up[v] = swap16(d0[v]);         // fetch rows 8-15

  const float ii[6] = { d0[0], d0[1], d0[2], up[0], up[1], up[2] };
  const float ff[6] = { d0[3], d0[4], d0[5], up[3], up[4], up[5] };
  const float oo[6] = { d0[6], d0[7], up[6], up[7], d1[0], d1[1] };
  const float gg[6] = { d1[2], d1[3], d1[4], d1[5], d1[6], d1[7] };
  #pragma unroll
  for (int j = 0; j < 6; ++j) {
    c[j] = ff[j] * c[j] + ii[j] * gg[j];
    h[j] = oo[j] * fast_tanh(c[j]);
  }
}

// B operand ([x_or_h(6); h(6)] zero-padded to K=32) in the 16-bit 32x16 B
// operand, contiguous k order (as A): lane n<16 holds K=0..15 of column n -> elements 0..11 = inputs;
// lanes 16-31 hold K=16..31 == 0. Upper-lane inputs are guaranteed 0.
DEVINL v16h pack_B(const float (&a)[6], const float (&b)[6]) {
  v16h B;
  #pragma unroll
  for (int i = 0; i < 16; ++i) B[i] = (_Float16)0.0f;
  #pragma unroll
  for (int j = 0; j < 6; ++j) {
    B[j]     = (_Float16)a[j];
    B[6 + j] = (_Float16)b[j];
  }
  return B;
}

__global__ __launch_bounds__(32)
void lstm2_wmma_kernel(const float* __restrict__ x,
                       const float* __restrict__ Wih0, const float* __restrict__ Whh0,
                       const float* __restrict__ bih0, const float* __restrict__ bhh0,
                       const float* __restrict__ Wih1, const float* __restrict__ Whh1,
                       const float* __restrict__ bih1, const float* __restrict__ bhh1,
                       float* __restrict__ out)
{
  const int  lane = threadIdx.x & 31;
  const bool lo   = lane < 16;
  const int  n    = blockIdx.x * 16 + (lane & 15);   // batch column for this lane

  // Resident weight/bias operands (constant over all 2048 timesteps).
  const v16h A00 = load_A(Wih0, Whh0, 0, lane);
  const v16h A01 = load_A(Wih0, Whh0, 1, lane);
  const v16h A10 = load_A(Wih1, Whh1, 0, lane);
  const v16h A11 = load_A(Wih1, Whh1, 1, lane);
  const v8f  C00 = load_bias(bih0, bhh0, 0, lane);
  const v8f  C01 = load_bias(bih0, bhh0, 1, lane);
  const v8f  C10 = load_bias(bih1, bhh1, 0, lane);
  const v8f  C11 = load_bias(bih1, bhh1, 1, lane);

  float h0[6], c0[6], h1[6], c1[6];
  #pragma unroll
  for (int j = 0; j < 6; ++j) { h0[j] = 0.f; c0[j] = 0.f; h1[j] = 0.f; c1[j] = 0.f; }

  const float* xrow = x + (size_t)n * kT * kD;
  float nx[6] = {0.f, 0.f, 0.f, 0.f, 0.f, 0.f};     // upper lanes stay 0 forever
  if (lo) {
    float2 a = *(const float2*)(xrow + 0);
    float2 b = *(const float2*)(xrow + 2);
    float2 c = *(const float2*)(xrow + 4);
    nx[0] = a.x; nx[1] = a.y; nx[2] = b.x; nx[3] = b.y; nx[4] = c.x; nx[5] = c.y;
  }

  for (int t = 0; t < kT; ++t) {
    float xv[6];
    #pragma unroll
    for (int j = 0; j < 6; ++j) xv[j] = nx[j];
    if (lo && (t + 1) < kT) {                        // register prefetch of x[t+1]
      const float* p = xrow + (size_t)(t + 1) * kD;
      float2 a = *(const float2*)(p + 0);
      float2 b = *(const float2*)(p + 2);
      float2 c = *(const float2*)(p + 4);
      nx[0] = a.x; nx[1] = a.y; nx[2] = b.x; nx[3] = b.y; nx[4] = c.x; nx[5] = c.y;
    }

    // ---- layer 0: gates^T = Wcat0 @ [x_t; h0]^T, bias via C operand ----
    {
      const v16h Bm = pack_B(xv, h0);
      v8f d0 = __builtin_amdgcn_wmma_f32_16x16x32_f16(false, A00, false, Bm,
                                                      (short)0, C00, false, false);
      v8f d1 = __builtin_amdgcn_wmma_f32_16x16x32_f16(false, A01, false, Bm,
                                                      (short)0, C01, false, false);
      step_update(d0, d1, h0, c0);
    }
    // ---- layer 1: gates^T = Wcat1 @ [h0; h1]^T ----
    {
      const v16h Bm = pack_B(h0, h1);
      v8f d0 = __builtin_amdgcn_wmma_f32_16x16x32_f16(false, A10, false, Bm,
                                                      (short)0, C10, false, false);
      v8f d1 = __builtin_amdgcn_wmma_f32_16x16x32_f16(false, A11, false, Bm,
                                                      (short)0, C11, false, false);
      step_update(d0, d1, h1, c1);
    }
  }

  // softmax over the 6 final layer-1 hidden values, per batch element
  if (lo) {
    float m = h1[0];
    #pragma unroll
    for (int j = 1; j < 6; ++j) m = fmaxf(m, h1[j]);
    float e[6], s = 0.f;
    #pragma unroll
    for (int j = 0; j < 6; ++j) {
      e[j] = __builtin_amdgcn_exp2f((h1[j] - m) * 1.44269504f);
      s += e[j];
    }
    float rs = 1.0f / s;
    #pragma unroll
    for (int j = 0; j < 6; ++j) out[(size_t)n * 6 + j] = e[j] * rs;
  }
}

extern "C" void kernel_launch(void* const* d_in, const int* in_sizes, int n_in,
                              void* d_out, int out_size, void* d_ws, size_t ws_size,
                              hipStream_t stream) {
  (void)in_sizes; (void)n_in; (void)out_size; (void)d_ws; (void)ws_size;
  const float* x    = (const float*)d_in[0];
  const float* Wih0 = (const float*)d_in[1];
  const float* Whh0 = (const float*)d_in[2];
  const float* bih0 = (const float*)d_in[3];
  const float* bhh0 = (const float*)d_in[4];
  const float* Wih1 = (const float*)d_in[5];
  const float* Whh1 = (const float*)d_in[6];
  const float* bih1 = (const float*)d_in[7];
  const float* bhh1 = (const float*)d_in[8];
  float* out = (float*)d_out;

  dim3 grid(kB / 16);   // 128 waves, one per workgroup, spread across WGPs
  dim3 block(32);       // wave32
  hipLaunchKernelGGL(lstm2_wmma_kernel, grid, block, 0, stream,
                     x, Wih0, Whh0, bih0, bhh0, Wih1, Whh1, bih1, bhh1, out);
}
